// QLinear_54004918779981
// MI455X (gfx1250) — 8-bit matrix operand order repaired and checked by reading, compiled, not run on hardware
//
#include <hip/hip_runtime.h>

typedef int v8i __attribute__((ext_vector_type(8)));

#define N_ROWS  65536
#define K_DIM   512
#define OUTF    512
#define BLK_M   128
#define BLK_N   128
#define THREADS 256

#define BF_WORDS (8 * 8 * 8 * 32)
#define AS_WORDS (BLK_M * 129)

__global__ __launch_bounds__(THREADS)
void qlinear_wmma_iu8_kernel(const float* __restrict__ x,
                             const int*   __restrict__ weight,
                             const int*   __restrict__ bias,
                             const float* __restrict__ Mp,
                             const float* __restrict__ xzp,
                             const float* __restrict__ wzp,
                             const float* __restrict__ yzp,
                             float*       __restrict__ out)
{
    __shared__ unsigned int lds[BF_WORDS + AS_WORDS];
    unsigned int* const Bf = lds;
    unsigned int* const As = lds + BF_WORDS;

    const int tid  = threadIdx.x;
    const int rb   = blockIdx.y * BLK_M;
    const int cb   = blockIdx.x * BLK_N;
    const int lane = tid & 31;
    const int nl   = lane & 15;
    const int h    = lane >> 4;

    const int xz = (int)rintf(*xzp);
    const int wz = (int)rintf(*wzp);

    {
        const int th = tid >> 7;
        const int tl = tid & 127;
        const float4* xg = (const float4*)(x + (size_t)(rb + th) * K_DIM) + tl;
        unsigned int* ap = &As[th * 129 + tl];
        #pragma unroll
        for (int i = 0; i < 64; ++i) {
            const float4 f = xg[i * 256];
            const int t0 = (int)f.x - xz;
            const int t1 = (int)f.y - xz;
            const int t2 = (int)f.z - xz;
            const int t3 = (int)f.w - xz;
            ap[i * 258] = (t0 & 0xff) | ((t1 & 0xff) << 8) |
                          ((t2 & 0xff) << 16) | ((unsigned)t3 << 24);
        }
    }

    {
        const int v  = (tid >> 5) & 7;
        const int vk = (v & 1) * 4 + h * 8 + (v >> 1) * 16;
        const int4* wg = (const int4*)(weight + (size_t)(cb + nl) * K_DIM + vk);
        unsigned int* bp = &Bf[tid];
        #pragma unroll
        for (int i = 0; i < 64; ++i) {
            const int ntg = i & 7;
            const int kc  = i >> 3;
            const int4 ww = wg[ntg * 2048 + kc * 16];
            const int t0 = ww.x - wz, t1 = ww.y - wz, t2 = ww.z - wz, t3 = ww.w - wz;
            bp[i * 256] = (t0 & 0xff) | ((t1 & 0xff) << 8) |
                          ((t2 & 0xff) << 16) | ((unsigned)t3 << 24);
        }
    }

    __syncthreads();

    const int wid = tid >> 5;
    const int wm  = wid & 3;
    const int wn  = wid >> 2;

    const unsigned int* ap0 = &As[(wm * 32 +  0 + nl) * 129 + h * 2];
    const unsigned int* ap1 = &As[(wm * 32 + 16 + nl) * 129 + h * 2];
    const unsigned int* bp  = &Bf[lane];

    v8i zero;
    #pragma unroll
    for (int j = 0; j < 8; ++j) zero[j] = 0;
    v8i acc[2][4];
    #pragma unroll
    for (int mt = 0; mt < 2; ++mt)
        #pragma unroll
        for (int nt = 0; nt < 4; ++nt) acc[mt][nt] = zero;

    #pragma unroll
    for (int kc = 0; kc < 8; ++kc) {
        v8i a0, a1, b[4];
        #pragma unroll
        for (int v = 0; v < 8; ++v) {
            const int kw = kc * 16 + (v >> 1) * 4 + (v & 1);
            a0[v] = (int)ap0[kw];
            a1[v] = (int)ap1[kw];
        }
        #pragma unroll
        for (int nt = 0; nt < 4; ++nt)
            #pragma unroll
            for (int v = 0; v < 8; ++v)
                b[nt][v] = (int)bp[(((kc * 8 + wn * 4 + nt) * 8) + v) * 32];
        #pragma unroll
        for (int nt = 0; nt < 4; ++nt) {
            acc[0][nt] = __builtin_amdgcn_wmma_i32_16x16x64_iu8(
                true, a0, true, b[nt], acc[0][nt], false, false);
            acc[1][nt] = __builtin_amdgcn_wmma_i32_16x16x64_iu8(
                true, a1, true, b[nt], acc[1][nt], false, false);
        }
    }

    const float Mv = *Mp;
    const float yz = *yzp;
    #pragma unroll
    for (int nt = 0; nt < 4; ++nt) {
        const int col = cb + wn * 64 + nt * 16 + nl;
        const float bz = (float)bias[col];
        #pragma unroll
        for (int mt = 0; mt < 2; ++mt) {
            const int rowb = rb + wm * 32 + mt * 16 + h * 8;
            #pragma unroll
            for (int j = 0; j < 8; ++j) {
                const float y = fmaf((float)acc[mt][nt][j] + bz, Mv, yz);
                __builtin_nontemporal_store(rintf(y),
                    &out[(size_t)(rowb + j) * OUTF + col]);
            }
        }
    }
}

extern "C" void kernel_launch(void* const* d_in, const int* in_sizes, int n_in,
                              void* d_out, int out_size, void* d_ws, size_t ws_size,
                              hipStream_t stream) {
    const float* x      = (const float*)d_in[0];
    const int*   weight = (const int*)  d_in[1];
    const int*   bias   = (const int*)  d_in[2];
    const float* Mp     = (const float*)d_in[3];
    const float* xzp    = (const float*)d_in[4];
    const float* wzp    = (const float*)d_in[5];
    const float* yzp    = (const float*)d_in[6];
    float* out = (float*)d_out;

    dim3 grid(OUTF / BLK_N, N_ROWS / BLK_M);
    qlinear_wmma_iu8_kernel<<<grid, THREADS, 0, stream>>>(
        x, weight, bias, Mp, xzp, wzp, yzp, out);
}
